// E29bSelectiveElmanCell_2147483648061
// MI455X (gfx1250) — compile-verified, matrix operand order checked by reading, not run on hardware
//
#include <hip/hip_runtime.h>
#include <hip/hip_bf16.h>

typedef __attribute__((ext_vector_type(16))) __bf16 v16bf;
typedef __attribute__((ext_vector_type(8)))  __bf16 v8bf;
typedef __attribute__((ext_vector_type(8)))  float  v8f;

#define Bdim 16
#define Tdim 512
#define Ddim 1024
#define Ndim 16
#define NBLOCKS 64

// ---------- helpers ----------
__device__ __forceinline__ __bf16 f2bf(float f) { return (__bf16)f; }  // native cvt

// Load a 16x32 bf16 tile (A: rows=M, or B: rows=N of row-major W so B[k][n]=W[n][k]).
// CDNA5 16-bit operand layout: lane&15 = row, lane>>4 selects K-groups {0..7,16..23} vs {8..15,24..31},
// two bf16 packed per VGPR in K order.
__device__ __forceinline__ v16bf load_tile_bf(const __bf16* __restrict__ base, int lda, int kb, int lane) {
  const int m  = lane & 15;
  const int kg = (lane >> 4) & 1;
  const __bf16* p = base + (long)m * lda + kb + kg * 8;
  v8bf lo = *(const v8bf*)(p);        // K = kb+kg*8 .. +7
  v8bf hi = *(const v8bf*)(p + 16);   // K = kb+16+kg*8 .. +7
  v16bf r;
#pragma unroll
  for (int i = 0; i < 8; ++i) { r[i] = lo[i]; r[i + 8] = hi[i]; }
  return r;
}

__device__ __forceinline__ v8f wmma_bf16(v16bf a, v16bf b, v8f c) {
  return __builtin_amdgcn_wmma_f32_16x16x32_bf16(false, a, false, b, (short)0, c, false, false);
}

// ---------- f32 -> bf16 conversion ----------
__global__ void __launch_bounds__(256) cvt_bf16(const float* __restrict__ s, __bf16* __restrict__ d, int n) {
  for (int i = blockIdx.x * 256 + threadIdx.x; i < n; i += gridDim.x * 256) d[i] = f2bf(s[i]);
}

// ---------- phase 1: xz = x @ W_xz.T ----------
// xbf:(B*T=8192, D) bf16, W_xz bf16:(2048, D) -> xproj f32 (cols 0..1023), z bf16 (cols 1024..2047)
// 2x2 register blocking per wave: 256 M-groups x 64 N-groups = 16384 wave units, 8 waves/block.
__global__ void __launch_bounds__(256) xz_gemm(const __bf16* __restrict__ xbf,
                                               const __bf16* __restrict__ wxzb,
                                               float* __restrict__ xproj,
                                               __bf16* __restrict__ zbf) {
  const int lane = threadIdx.x & 31;
  const int wave = threadIdx.x >> 5;
  const int unit = blockIdx.x * 8 + wave;   // 16384 units
  const int ng2  = unit & 63;               // N group of 2 tiles
  const int mg2  = unit >> 6;               // M group of 2 tiles
  const int mt0 = mg2 * 2, nt0 = ng2 * 2;

  const __bf16* A0 = xbf  + (long)(mt0    ) * 16 * Ddim;
  const __bf16* A1 = xbf  + (long)(mt0 + 1) * 16 * Ddim;
  const __bf16* B0 = wxzb + (long)(nt0    ) * 16 * Ddim;
  const __bf16* B1 = wxzb + (long)(nt0 + 1) * 16 * Ddim;

  v8f c00 = {}, c01 = {}, c10 = {}, c11 = {};
#pragma unroll 2
  for (int kb = 0; kb < Ddim; kb += 32) {
    v16bf a0 = load_tile_bf(A0, Ddim, kb, lane);
    v16bf a1 = load_tile_bf(A1, Ddim, kb, lane);
    v16bf b0 = load_tile_bf(B0, Ddim, kb, lane);
    v16bf b1 = load_tile_bf(B1, Ddim, kb, lane);
    c00 = wmma_bf16(a0, b0, c00);
    c01 = wmma_bf16(a0, b1, c01);
    c10 = wmma_bf16(a1, b0, c10);
    c11 = wmma_bf16(a1, b1, c11);
  }

  const int nloc = lane & 15;
  const int mh   = lane >> 4;
#pragma unroll
  for (int q = 0; q < 4; ++q) {
    const int mt = mt0 + (q >> 1);
    const int nt = nt0 + (q & 1);
    v8f acc = (q == 0) ? c00 : (q == 1) ? c01 : (q == 2) ? c10 : c11;
    const int ng = nt * 16 + nloc;
#pragma unroll
    for (int r = 0; r < 8; ++r) {
      const int mg = mt * 16 + r + 8 * mh;
      if (ng < Ddim) xproj[(long)mg * Ddim + ng] = acc[r];
      else           zbf[(long)mg * Ddim + (ng - Ddim)] = f2bf(acc[r]);
    }
  }
}

// ---------- grid-wide barrier (monotonic counting; counter reset per launch) ----------
__device__ __forceinline__ void gsync(unsigned* bar, unsigned& ph) {
  __syncthreads();
  if (threadIdx.x == 0) {
    __threadfence();
    atomicAdd(bar, 1u);
    ph++;
    const unsigned target = (unsigned)NBLOCKS * ph;
    while (*((volatile unsigned*)bar) < target) { __builtin_amdgcn_s_sleep(1); }
    __threadfence();
  }
  __syncthreads();
}

// ---------- shared helpers for the scan kernel ----------
__device__ __forceinline__ float block_reduce_tile(v8f acc, float* red, int wave, int lane,
                                                   int& m, int& nloc) {
#pragma unroll
  for (int r = 0; r < 8; ++r) red[wave * 256 + lane * 8 + r] = acc[r];
  __syncthreads();
  const int i  = threadIdx.x;
  const int li = i >> 3;
  const int r  = i & 7;
  float s = 0.f;
#pragma unroll
  for (int w = 0; w < 8; ++w) s += red[w * 256 + li * 8 + r];
  m    = r + ((li >> 4) << 3);
  nloc = li & 15;
  return s;
}

__device__ __forceinline__ void attn_weights(const float* __restrict__ q, const float* __restrict__ tp,
                                             float* s_p, float* s_sc, float* s_attn) {
  const int tid = threadIdx.x;
  const int n = tid >> 4, sub = tid & 15;
  float p = 0.f;
  for (int d = sub; d < Ddim; d += 16) p += q[d] * tp[n * Ddim + d];
  s_p[tid] = p;
  __syncthreads();
  if (sub == 0) {
    float s = 0.f;
#pragma unroll
    for (int j = 0; j < 16; ++j) s += s_p[(n << 4) + j];
    s_sc[n] = s * 0.03125f;  // 1/sqrt(1024)
  }
  __syncthreads();
  if (tid == 0) {
    float mx = s_sc[0];
    for (int j = 1; j < 16; ++j) mx = fmaxf(mx, s_sc[j]);
    float den = 0.f;
    for (int j = 0; j < 16; ++j) { float e = __expf(s_sc[j] - mx); s_attn[j] = e; den += e; }
    const float inv = 1.f / den;
    for (int j = 0; j < 16; ++j) s_attn[j] *= inv;
  }
  __syncthreads();
}

// ---------- phase 2: persistent sequential scan ----------
__global__ void __launch_bounds__(256) scan_kernel(
    const float*  __restrict__ xproj,   // (B,T,D) f32
    const __bf16* __restrict__ zbf,     // (B,T,D) bf16
    const __bf16* __restrict__ whb,     // (D,D)
    const __bf16* __restrict__ wwrb,    // (D,D)
    const __bf16* __restrict__ wgb,     // (D,3D)
    const float*  __restrict__ b_h,     // (D)
    const float*  __restrict__ h_init,  // (B,D) f32
    float* __restrict__ tape,           // (B,N,D) f32 -- lives in d_out tail
    float* __restrict__ out,            // (B,T,D) f32
    float* __restrict__ hf32,           // (B,D)
    __bf16* __restrict__ hbf,           // 2 x (B,D) double-buffered bf16 h
    float* __restrict__ rv,             // (B,D) read_val f32
    __bf16* __restrict__ rvbf,          // (B,D) read_val bf16
    float* __restrict__ wvv,            // (B,D) write_val f32
    unsigned* __restrict__ bar) {
  __shared__ float red[2048];
  __shared__ float s_p[256];
  __shared__ float s_sc[16];
  __shared__ float s_attn[16];

  const int lane = threadIdx.x & 31;
  const int wave = threadIdx.x >> 5;
  const int nt   = blockIdx.x;          // one 16-col output tile per block (64 tiles = D)
  unsigned ph = 0;

  // ---- prologue: read attention for t=0 (blocks 0..15, one per batch) ----
  if (blockIdx.x < Bdim) {
    const int b = blockIdx.x;
    float* tp = tape + (long)b * Ndim * Ddim;
    attn_weights(h_init + (long)b * Ddim, tp, s_p, s_sc, s_attn);
    for (int d = threadIdx.x; d < Ddim; d += 256) {
      float a = 0.f;
#pragma unroll
      for (int n = 0; n < 16; ++n) a += s_attn[n] * tp[n * Ddim + d];
      rv[(long)b * Ddim + d]   = a;
      rvbf[(long)b * Ddim + d] = f2bf(a);
    }
  }

  for (int t = 0; t < Tdim; ++t) {
    gsync(bar, ph);

    // ---- Phase B: h_new = tanh(xp[t] + h @ W_h.T + read_val + b_h) ----
    {
      const __bf16* Acur  = hbf + (long)(t & 1) * Bdim * Ddim;
      __bf16*       Anext = (__bf16*)hbf + (long)((t + 1) & 1) * Bdim * Ddim;
      const __bf16* Bw = whb + (long)nt * 16 * Ddim;
      v8f acc = {};
      const int k0 = wave * 128;
#pragma unroll
      for (int kk = 0; kk < 128; kk += 32) {
        __builtin_prefetch(Bw + (long)(lane & 15) * Ddim + k0 + kk + 32, 0, 3);
        v16bf a = load_tile_bf(Acur, Ddim, k0 + kk, lane);
        v16bf b = load_tile_bf(Bw, Ddim, k0 + kk, lane);
        acc = wmma_bf16(a, b, acc);
      }
      int m, nloc;
      float s = block_reduce_tile(acc, red, wave, lane, m, nloc);
      const int ng = nt * 16 + nloc;
      float val = tanhf(xproj[((long)m * Tdim + t) * Ddim + ng] + rv[(long)m * Ddim + ng] + b_h[ng] + s);
      hf32[(long)m * Ddim + ng]  = val;
      Anext[(long)m * Ddim + ng] = f2bf(val);
      __syncthreads();
    }
    gsync(bar, ph);

    // ---- Phase C: write_val = h_new @ W_write.T ; out = h_new * silu(gate_in @ W_gate.T) ----
    {
      const __bf16* Ah = hbf + (long)((t + 1) & 1) * Bdim * Ddim;
      // write_val GEMM
      {
        const __bf16* Bw = wwrb + (long)nt * 16 * Ddim;
        v8f acc = {};
        const int k0 = wave * 128;
#pragma unroll
        for (int kk = 0; kk < 128; kk += 32) {
          __builtin_prefetch(Bw + (long)(lane & 15) * Ddim + k0 + kk + 32, 0, 3);
          v16bf a = load_tile_bf(Ah, Ddim, k0 + kk, lane);
          v16bf b = load_tile_bf(Bw, Ddim, k0 + kk, lane);
          acc = wmma_bf16(a, b, acc);
        }
        int m, nloc;
        float s = block_reduce_tile(acc, red, wave, lane, m, nloc);
        wvv[(long)m * Ddim + (nt * 16 + nloc)] = s;
        __syncthreads();
      }
      // gate GEMM: K=3072, segmented A = [z_t | read_val | h_new] (all bf16)
      {
        const __bf16* Bg = wgb + (long)nt * 16 * (3 * Ddim);
        v8f acc = {};
        const int k0 = wave * 384;
#pragma unroll
        for (int kk = 0; kk < 384; kk += 32) {
          const int kb = k0 + kk;
          __builtin_prefetch(Bg + (long)(lane & 15) * (3 * Ddim) + kb + 32, 0, 3);
          const __bf16* Ab; int la, ko;
          if (kb < Ddim)            { Ab = zbf + (long)t * Ddim; la = Tdim * Ddim; ko = kb; }
          else if (kb < 2 * Ddim)   { Ab = rvbf; la = Ddim; ko = kb - Ddim; }
          else                      { Ab = Ah;   la = Ddim; ko = kb - 2 * Ddim; }
          v16bf a = load_tile_bf(Ab, la, ko, lane);
          v16bf b = load_tile_bf(Bg, 3 * Ddim, kb, lane);
          acc = wmma_bf16(a, b, acc);
        }
        int m, nloc;
        float g = block_reduce_tile(acc, red, wave, lane, m, nloc);
        const int ng = nt * 16 + nloc;
        float gate = g / (1.f + __expf(-g));                 // silu
        out[((long)m * Tdim + t) * Ddim + ng] = hf32[(long)m * Ddim + ng] * gate;
        __syncthreads();
      }
    }
    gsync(bar, ph);

    // ---- Phase D: write attention + tape update + read attention for t+1 (blocks 0..15) ----
    if (blockIdx.x < Bdim) {
      const int b = blockIdx.x;
      float* tp = tape + (long)b * Ndim * Ddim;
      const float* w = wvv + (long)b * Ddim;
      attn_weights(w, tp, s_p, s_sc, s_attn);
      for (int idx = threadIdx.x; idx < Ndim * Ddim; idx += 256) {
        const int d = idx & (Ddim - 1);
        const float a = s_attn[idx >> 10];
        tp[idx] = tp[idx] * (1.f - a) + w[d] * a;
      }
      __syncthreads();
      attn_weights(hf32 + (long)b * Ddim, tp, s_p, s_sc, s_attn);
      for (int d = threadIdx.x; d < Ddim; d += 256) {
        float a = 0.f;
#pragma unroll
        for (int n = 0; n < 16; ++n) a += s_attn[n] * tp[n * Ddim + d];
        rv[(long)b * Ddim + d]   = a;
        rvbf[(long)b * Ddim + d] = f2bf(a);
      }
    }
  }
}

// ---------- launcher ----------
extern "C" void kernel_launch(void* const* d_in, const int* in_sizes, int n_in,
                              void* d_out, int out_size, void* d_ws, size_t ws_size,
                              hipStream_t stream) {
  const float* x         = (const float*)d_in[0];
  const float* tape_init = (const float*)d_in[1];
  const float* h_init    = (const float*)d_in[2];
  const float* W_h       = (const float*)d_in[3];
  const float* W_xz      = (const float*)d_in[4];
  const float* b_h       = (const float*)d_in[5];
  const float* W_wr      = (const float*)d_in[6];
  const float* W_g       = (const float*)d_in[7];

  float* out      = (float*)d_out;                              // (B,T,D)
  float* out_tape = out + (size_t)Bdim * Tdim * Ddim;           // (B,N,D)

  char* w = (char*)d_ws;
  size_t off = 0;
  auto alloc = [&](size_t bytes) -> void* {
    void* p = w + off; off += (bytes + 255) & ~(size_t)255; return p;
  };
  unsigned* bar  = (unsigned*)alloc(256);
  __bf16* whb   = (__bf16*)alloc((size_t)Ddim * Ddim * 2);
  __bf16* wwrb  = (__bf16*)alloc((size_t)Ddim * Ddim * 2);
  __bf16* wgb   = (__bf16*)alloc((size_t)Ddim * 3 * Ddim * 2);
  __bf16* wxzb  = (__bf16*)alloc((size_t)2 * Ddim * Ddim * 2);
  __bf16* xbf   = (__bf16*)alloc((size_t)Bdim * Tdim * Ddim * 2);
  float*  xproj = (float*)alloc((size_t)Bdim * Tdim * Ddim * 4);
  __bf16* zbf   = (__bf16*)alloc((size_t)Bdim * Tdim * Ddim * 2);
  __bf16* hbf   = (__bf16*)alloc((size_t)2 * Bdim * Ddim * 2);
  float*  hf32  = (float*)alloc((size_t)Bdim * Ddim * 4);
  float*  rv    = (float*)alloc((size_t)Bdim * Ddim * 4);
  __bf16* rvbf  = (__bf16*)alloc((size_t)Bdim * Ddim * 2);
  float*  wvv   = (float*)alloc((size_t)Bdim * Ddim * 4);

  hipMemsetAsync(bar, 0, 256, stream);
  hipMemcpyAsync(out_tape, tape_init, (size_t)Bdim * Ndim * Ddim * sizeof(float),
                 hipMemcpyDeviceToDevice, stream);

  // weight / activation / initial-state conversion to bf16
  {
    int n;
    n = Ddim * Ddim;            cvt_bf16<<<(n + 255) / 256, 256, 0, stream>>>(W_h,  whb,  n);
    n = Ddim * Ddim;            cvt_bf16<<<(n + 255) / 256, 256, 0, stream>>>(W_wr, wwrb, n);
    n = Ddim * 3 * Ddim;        cvt_bf16<<<(n + 255) / 256, 256, 0, stream>>>(W_g,  wgb,  n);
    n = 2 * Ddim * Ddim;        cvt_bf16<<<(n + 255) / 256, 256, 0, stream>>>(W_xz, wxzb, n);
    n = Bdim * Tdim * Ddim;     cvt_bf16<<<(n + 255) / 256, 256, 0, stream>>>(x,    xbf,  n);
    n = Bdim * Ddim;            cvt_bf16<<<(n + 255) / 256, 256, 0, stream>>>(h_init, hbf, n); // hbf[0]
  }

  // phase 1: xz precompute, 2x2 register-blocked WMMA (16384 wave units / 8 waves per block)
  xz_gemm<<<2048, 256, 0, stream>>>(xbf, wxzb, xproj, zbf);

  // phase 2: persistent recurrent scan
  scan_kernel<<<NBLOCKS, 256, 0, stream>>>(xproj, zbf, whb, wwrb, wgb, b_h, h_init,
                                           out_tape, out, hf32, hbf, rv, rvbf, wvv, bar);
}
